// TransLayer_78134045048857
// MI455X (gfx1250) — compile-verified, matrix operand order checked by reading, not run on hardware
//
#include <hip/hip_runtime.h>

typedef __bf16 bf16_t;
typedef __attribute__((ext_vector_type(16))) __bf16 v16bf;
typedef __attribute__((ext_vector_type(8)))  float  v8f;

#define BATCH 4
#define SEQ   8192
#define DIM   512
#define HEADS 8
#define DHEAD 64
#define LAND  256
#define BH    (BATCH*HEADS)      // 32
#define TOK   (BATCH*SEQ)        // 32768
#define KSIZE 33

// ---------------- WMMA helpers (CDNA5 wave32) ----------------
// Fragment loader for A or B operand of v_wmma_f32_16x16x32_bf16.
// Source: row-major [rows, ld] bf16, tile at (r0, k0), K-slab of 32.
static __device__ __forceinline__ v16bf load_frag(const bf16_t* p, int ld, int r0, int k0) {
  int lane = threadIdx.x & 31;
  int r    = r0 + (lane & 15);
  int kh   = (lane >> 4) << 3;            // 0 or 8
  const bf16_t* base = p + (size_t)r * ld + k0;
  union { unsigned u[8]; v16bf v; } f;
#pragma unroll
  for (int i = 0; i < 8; ++i) {
    int kk = ((i & 4) << 2) + kh + ((i & 3) << 1);  // {0,16} + {0,8} + 2*(i&3)
    f.u[i] = *(const unsigned*)(base + kk);
  }
  return f.v;
}

static __device__ __forceinline__ v8f wmma_bf16(v16bf a, v16bf b, v8f c) {
  return __builtin_amdgcn_wmma_f32_16x16x32_bf16(false, a, false, b, (short)0, c, false, false);
}

// C-tile layout: VGPR i, lane L -> row = i + 8*(L>>4), col = L&15.

// ---------------- weight convert + scalar init ----------------
__global__ void k_convert(const float* __restrict__ wqkv, const float* __restrict__ wout,
                          bf16_t* __restrict__ wqkvT, bf16_t* __restrict__ woutT,
                          unsigned* __restrict__ rowmax) {
  int idx = blockIdx.x * 256 + threadIdx.x;
  if (idx == 0) *rowmax = 0u;
  if (idx < DIM * 3 * DIM) {                       // [512][1536] -> [1536][512]
    int r = idx / (3 * DIM), c = idx % (3 * DIM);
    wqkvT[(size_t)c * DIM + r] = (bf16_t)wqkv[idx];
  }
  if (idx < DIM * DIM) {                           // [512][512] -> transposed
    int r = idx / DIM, c = idx % DIM;
    woutT[(size_t)c * DIM + r] = (bf16_t)wout[idx];
  }
}

// ---------------- LayerNorm -> xn bf16 ----------------
__global__ void k_layernorm(const float* __restrict__ x, const float* __restrict__ gamma,
                            const float* __restrict__ beta, bf16_t* __restrict__ xn) {
  int lane = threadIdx.x & 31;
  int row  = blockIdx.x * 8 + (threadIdx.x >> 5);
  const float* xr = x + (size_t)row * DIM;
  float vals[16], s = 0.f, s2 = 0.f;
#pragma unroll
  for (int i = 0; i < 16; ++i) { float v = xr[lane + 32 * i]; vals[i] = v; s += v; s2 += v * v; }
#pragma unroll
  for (int m = 16; m >= 1; m >>= 1) { s += __shfl_xor(s, m, 32); s2 += __shfl_xor(s2, m, 32); }
  float mu  = s * (1.f / DIM);
  float inv = rsqrtf(s2 * (1.f / DIM) - mu * mu + 1e-5f);
  bf16_t* o = xn + (size_t)row * DIM;
#pragma unroll
  for (int i = 0; i < 16; ++i) {
    int c = lane + 32 * i;
    o[c] = (bf16_t)(((vals[i] - mu) * inv) * gamma[c] + beta[c]);
  }
}

// ---------------- QKV GEMM: 32x32 C per wave, scatter epilogue ----------------
__global__ void k_gemm_qkv(const bf16_t* __restrict__ A, const bf16_t* __restrict__ B,
                           bf16_t* __restrict__ q, bf16_t* __restrict__ k,
                           bf16_t* __restrict__ v, bf16_t* __restrict__ vT) {
  int m0 = blockIdx.x * 32, n0 = blockIdx.y * 32;
  v8f acc[2][2] = {};
  for (int k0 = 0; k0 < DIM; k0 += 32) {
    v16bf a0 = load_frag(A, DIM, m0, k0),      a1 = load_frag(A, DIM, m0 + 16, k0);
    v16bf b0 = load_frag(B, DIM, n0, k0),      b1 = load_frag(B, DIM, n0 + 16, k0);
    acc[0][0] = wmma_bf16(a0, b0, acc[0][0]);  acc[0][1] = wmma_bf16(a0, b1, acc[0][1]);
    acc[1][0] = wmma_bf16(a1, b0, acc[1][0]);  acc[1][1] = wmma_bf16(a1, b1, acc[1][1]);
  }
  int lane = threadIdx.x & 31;
  int rsel = (lane >> 4) << 3;
#pragma unroll
  for (int mi = 0; mi < 2; ++mi)
#pragma unroll
    for (int ni = 0; ni < 2; ++ni) {
      int col = n0 + ni * 16 + (lane & 15);
      int which = col >> 9;          // 0=q 1=k 2=v
      int cc = col & 511;
      int h = cc >> 6, d = cc & 63;
#pragma unroll
      for (int i = 0; i < 8; ++i) {
        int row = m0 + mi * 16 + i + rsel;     // token id
        int b_ = row >> 13, nn = row & (SEQ - 1);
        size_t bh = (size_t)b_ * HEADS + h;
        size_t idx = (bh * SEQ + nn) * DHEAD + d;
        float val = acc[mi][ni][i];
        if (which == 0)      q[idx] = (bf16_t)(val * 0.125f);   // dhead^-0.5
        else if (which == 1) k[idx] = (bf16_t)val;
        else { v[idx] = (bf16_t)val; vT[(bh * DHEAD + d) * SEQ + nn] = (bf16_t)val; }
      }
    }
}

// ---------------- landmark means ----------------
__global__ void k_landmarks(const bf16_t* __restrict__ q, const bf16_t* __restrict__ k,
                            bf16_t* __restrict__ ql, bf16_t* __restrict__ kl) {
  int idx = blockIdx.x * 256 + threadIdx.x;     // bh*256*64
  if (idx >= BH * LAND * DHEAD) return;
  int d = idx & 63, m = (idx >> 6) & 255, bh = idx >> 14;
  size_t base = ((size_t)bh * SEQ + (size_t)m * 32) * DHEAD + d;
  float sq = 0.f, sk = 0.f;
#pragma unroll 8
  for (int i = 0; i < 32; ++i) { sq += (float)q[base + (size_t)i * DHEAD]; sk += (float)k[base + (size_t)i * DHEAD]; }
  ql[idx] = (bf16_t)(sq * (1.f / 32.f));
  kl[idx] = (bf16_t)(sk * (1.f / 32.f));
}

// ---------------- sim2 = softmax(q_l k_l^T) ----------------
__global__ void k_sim2(const bf16_t* __restrict__ ql, const bf16_t* __restrict__ kl,
                       float* __restrict__ a2f, bf16_t* __restrict__ a2b) {
  int bh = blockIdx.y, m0 = blockIdx.x * 16;
  const bf16_t* A = ql + (size_t)bh * LAND * DHEAD;
  const bf16_t* B = kl + (size_t)bh * LAND * DHEAD;
  v16bf a0 = load_frag(A, DHEAD, m0, 0), a1 = load_frag(A, DHEAD, m0, 32);
  v8f s[16];
#pragma unroll
  for (int t = 0; t < 16; ++t) {
    v8f z = {};
    z = wmma_bf16(a0, load_frag(B, DHEAD, t * 16, 0), z);
    z = wmma_bf16(a1, load_frag(B, DHEAD, t * 16, 32), z);
    s[t] = z;
  }
  int lane = threadIdx.x & 31;
#pragma unroll
  for (int i = 0; i < 8; ++i) {
    float mx = -1e30f;
#pragma unroll
    for (int t = 0; t < 16; ++t) mx = fmaxf(mx, s[t][i]);
#pragma unroll
    for (int m = 1; m < 16; m <<= 1) mx = fmaxf(mx, __shfl_xor(mx, m, 32));
    float sm = 0.f;
#pragma unroll
    for (int t = 0; t < 16; ++t) { float p = __expf(s[t][i] - mx); s[t][i] = p; sm += p; }
#pragma unroll
    for (int m = 1; m < 16; m <<= 1) sm += __shfl_xor(sm, m, 32);
    float inv = 1.f / sm;
    int row = m0 + i + ((lane >> 4) << 3);
#pragma unroll
    for (int t = 0; t < 16; ++t) {
      int col = t * 16 + (lane & 15);
      float p = s[t][i] * inv;
      size_t o = ((size_t)bh * LAND + row) * LAND + col;
      a2f[o] = p; a2b[o] = (bf16_t)p;
    }
  }
}

// ---------------- global max of column sums (pinv init scalar; row sums are 1) ----------------
__global__ void k_colmax(const float* __restrict__ a2f, unsigned* __restrict__ rowmax) {
  int bh = blockIdx.x, j = threadIdx.x;
  const float* p = a2f + (size_t)bh * LAND * LAND + j;
  float s = 0.f;
  for (int i = 0; i < LAND; ++i) s += p[(size_t)i * LAND];
  atomicMax(rowmax, __float_as_uint(s));   // positive floats order as uints
}

// ---------------- z0 = a2^T / rowmax ----------------
__global__ void k_z0(const float* __restrict__ a2f, const unsigned* __restrict__ rowmax,
                     bf16_t* __restrict__ z, bf16_t* __restrict__ zT) {
  int idx = blockIdx.x * 256 + threadIdx.x;
  if (idx >= BH * LAND * LAND) return;
  int bh = idx >> 16, i = (idx >> 8) & 255, j = idx & 255;
  float inv = 1.f / __uint_as_float(*rowmax);
  z[idx]  = (bf16_t)(a2f[((size_t)bh << 16) + (size_t)j * LAND + i] * inv);  // a2^T
  zT[idx] = (bf16_t)(a2f[idx] * inv);
}

// ---------------- generic batched NT GEMM, 32x32 C per wave, (scale,diag) epilogues ----------------
__global__ void k_gemm_nt(const bf16_t* __restrict__ A, const bf16_t* __restrict__ B,
                          int M, int N, int K, long long sA, long long sB,
                          bf16_t* __restrict__ Cb, float scb, float dcb,
                          bf16_t* __restrict__ Ct, float sct, float dct) {
  int b_ = blockIdx.z;
  const bf16_t* Ab = A + (size_t)b_ * sA;
  const bf16_t* Bb = B + (size_t)b_ * sB;
  int m0 = blockIdx.x * 32, n0 = blockIdx.y * 32;
  v8f acc[2][2] = {};
  for (int k0 = 0; k0 < K; k0 += 32) {
    v16bf a0 = load_frag(Ab, K, m0, k0),      a1 = load_frag(Ab, K, m0 + 16, k0);
    v16bf b0 = load_frag(Bb, K, n0, k0),      b1 = load_frag(Bb, K, n0 + 16, k0);
    acc[0][0] = wmma_bf16(a0, b0, acc[0][0]); acc[0][1] = wmma_bf16(a0, b1, acc[0][1]);
    acc[1][0] = wmma_bf16(a1, b0, acc[1][0]); acc[1][1] = wmma_bf16(a1, b1, acc[1][1]);
  }
  int lane = threadIdx.x & 31;
  int rsel = (lane >> 4) << 3;
  size_t cbase = (size_t)b_ * M * N;
#pragma unroll
  for (int mi = 0; mi < 2; ++mi)
#pragma unroll
    for (int ni = 0; ni < 2; ++ni) {
      int col = n0 + ni * 16 + (lane & 15);
#pragma unroll
      for (int i = 0; i < 8; ++i) {
        int row = m0 + mi * 16 + i + rsel;
        float v = acc[mi][ni][i];
        float dg = (row == col) ? 1.f : 0.f;
        if (Cb) Cb[cbase + (size_t)row * N + col] = (bf16_t)(scb * v + dcb * dg);
        if (Ct) Ct[cbase + (size_t)col * M + row] = (bf16_t)(sct * v + dct * dg);
      }
    }
}

// ---------------- flash-style a3@v: 256 queries (q_l) over n keys; writes (a3v)^T ----------------
__global__ void k_flash(const bf16_t* __restrict__ ql, const bf16_t* __restrict__ k,
                        const bf16_t* __restrict__ vT, bf16_t* __restrict__ a3vT) {
  __shared__ bf16_t Pl[16 * 32];
  int bh = blockIdx.y, q0 = blockIdx.x * 16;
  const bf16_t* Aq = ql + (size_t)bh * LAND * DHEAD;
  const bf16_t* Kb = k  + (size_t)bh * SEQ * DHEAD;
  const bf16_t* Vt = vT + (size_t)bh * DHEAD * SEQ;
  v16bf aq0 = load_frag(Aq, DHEAD, q0, 0), aq1 = load_frag(Aq, DHEAD, q0, 32);
  int lane = threadIdx.x & 31;
  int rsel = (lane >> 4) << 3;
  float mrow[8], lrow[8];
  v8f O[4] = {};
#pragma unroll
  for (int i = 0; i < 8; ++i) { mrow[i] = -1e30f; lrow[i] = 0.f; }
  for (int kb = 0; kb < SEQ; kb += 32) {
    v8f s0 = {}, s1 = {};
    s0 = wmma_bf16(aq0, load_frag(Kb, DHEAD, kb, 0), s0);
    s0 = wmma_bf16(aq1, load_frag(Kb, DHEAD, kb, 32), s0);
    s1 = wmma_bf16(aq0, load_frag(Kb, DHEAD, kb + 16, 0), s1);
    s1 = wmma_bf16(aq1, load_frag(Kb, DHEAD, kb + 16, 32), s1);
#pragma unroll
    for (int i = 0; i < 8; ++i) {
      float t = fmaxf(s0[i], s1[i]);
#pragma unroll
      for (int m = 1; m < 16; m <<= 1) t = fmaxf(t, __shfl_xor(t, m, 32));
      float mn = fmaxf(mrow[i], t);
      float scale = __expf(mrow[i] - mn);
      mrow[i] = mn;
      float p0 = __expf(s0[i] - mn), p1 = __expf(s1[i] - mn);
      int row = i + rsel;
      Pl[row * 32 + (lane & 15)]      = (bf16_t)p0;
      Pl[row * 32 + (lane & 15) + 16] = (bf16_t)p1;
      float ps = p0 + p1;
#pragma unroll
      for (int m = 1; m < 16; m <<= 1) ps += __shfl_xor(ps, m, 32);
      lrow[i] = lrow[i] * scale + ps;
#pragma unroll
      for (int t4 = 0; t4 < 4; ++t4) O[t4][i] *= scale;
    }
    v16bf pa = load_frag(Pl, 32, 0, 0);
#pragma unroll
    for (int t4 = 0; t4 < 4; ++t4)
      O[t4] = wmma_bf16(pa, load_frag(Vt, SEQ, t4 * 16, kb), O[t4]);
  }
#pragma unroll
  for (int i = 0; i < 8; ++i) {
    int row = q0 + i + rsel;
    float inv = 1.f / lrow[i];
#pragma unroll
    for (int t4 = 0; t4 < 4; ++t4) {
      int d = (lane & 15) + 16 * t4;
      a3vT[(size_t)bh * DHEAD * LAND + (size_t)d * LAND + row] = (bf16_t)(O[t4][i] * inv);
    }
  }
}

// ---------------- fused sim1 -> softmax -> @W : out_head f32 [bh,n,64] ----------------
__global__ void k_attn(const bf16_t* __restrict__ q, const bf16_t* __restrict__ kl,
                       const bf16_t* __restrict__ WT, float* __restrict__ ohf) {
  __shared__ bf16_t Pl[16 * 32];
  int bh = blockIdx.y, n0 = blockIdx.x * 16;
  const bf16_t* Aq = q  + (size_t)bh * SEQ * DHEAD;
  const bf16_t* Kl = kl + (size_t)bh * LAND * DHEAD;
  const bf16_t* Wt = WT + (size_t)bh * DHEAD * LAND;   // [64][256]
  v16bf a0 = load_frag(Aq, DHEAD, n0, 0), a1 = load_frag(Aq, DHEAD, n0, 32);
  v8f s[16];
#pragma unroll
  for (int t = 0; t < 16; ++t) {
    v8f z = {};
    z = wmma_bf16(a0, load_frag(Kl, DHEAD, t * 16, 0), z);
    z = wmma_bf16(a1, load_frag(Kl, DHEAD, t * 16, 32), z);
    s[t] = z;
  }
  int lane = threadIdx.x & 31;
  int rsel = (lane >> 4) << 3;
  float inv[8];
#pragma unroll
  for (int i = 0; i < 8; ++i) {
    float mx = -1e30f;
#pragma unroll
    for (int t = 0; t < 16; ++t) mx = fmaxf(mx, s[t][i]);
#pragma unroll
    for (int m = 1; m < 16; m <<= 1) mx = fmaxf(mx, __shfl_xor(mx, m, 32));
    float sm = 0.f;
#pragma unroll
    for (int t = 0; t < 16; ++t) { float p = __expf(s[t][i] - mx); s[t][i] = p; sm += p; }
#pragma unroll
    for (int m = 1; m < 16; m <<= 1) sm += __shfl_xor(sm, m, 32);
    inv[i] = 1.f / sm;
  }
  v8f O[4] = {};
  for (int kb = 0; kb < 8; ++kb) {            // 256 landmarks in K-slabs of 32
#pragma unroll
    for (int j = 0; j < 2; ++j)
#pragma unroll
      for (int i = 0; i < 8; ++i)
        Pl[(i + rsel) * 32 + (lane & 15) + 16 * j] = (bf16_t)s[2 * kb + j][i];
    v16bf pa = load_frag(Pl, 32, 0, 0);
#pragma unroll
    for (int u = 0; u < 4; ++u)
      O[u] = wmma_bf16(pa, load_frag(Wt, LAND, u * 16, kb * 32), O[u]);
  }
#pragma unroll
  for (int i = 0; i < 8; ++i) {
    int row = n0 + i + rsel;
#pragma unroll
    for (int u = 0; u < 4; ++u) {
      int d = (lane & 15) + 16 * u;
      ohf[((size_t)bh * SEQ + row) * DHEAD + d] = O[u][i] * inv[i];
    }
  }
}

// ---------------- depthwise conv residual + head-merge to bf16 [b,n,512] ----------------
__global__ void k_conv(const float* __restrict__ ohf, const bf16_t* __restrict__ v,
                       const float* __restrict__ wconv, bf16_t* __restrict__ oh) {
  size_t idx = (size_t)blockIdx.x * 256 + threadIdx.x;
  if (idx >= (size_t)BH * SEQ * DHEAD) return;
  int d = idx & 63;
  size_t r = idx >> 6;
  int nn = (int)(r & (SEQ - 1));
  int bh = (int)(r >> 13);
  int h = bh & 7, b_ = bh >> 3;
  float acc = ohf[idx];
  const bf16_t* vb = v + ((size_t)bh * SEQ) * DHEAD + d;
  const float* w = wconv + h * KSIZE;
#pragma unroll 8
  for (int t = 0; t < KSIZE; ++t) {
    int src = nn + t - 16;
    if (src >= 0 && src < SEQ) acc += w[t] * (float)vb[(size_t)src * DHEAD];
  }
  oh[((size_t)b_ * SEQ + nn) * DIM + h * DHEAD + d] = (bf16_t)acc;
}

// ---------------- output projection + bias + residual, 32x32 C per wave ----------------
__global__ void k_gemm_out(const bf16_t* __restrict__ A, const bf16_t* __restrict__ B,
                           const float* __restrict__ bias, const float* __restrict__ x,
                           float* __restrict__ y) {
  int m0 = blockIdx.x * 32, n0 = blockIdx.y * 32;
  v8f acc[2][2] = {};
  for (int k0 = 0; k0 < DIM; k0 += 32) {
    v16bf a0 = load_frag(A, DIM, m0, k0),      a1 = load_frag(A, DIM, m0 + 16, k0);
    v16bf b0 = load_frag(B, DIM, n0, k0),      b1 = load_frag(B, DIM, n0 + 16, k0);
    acc[0][0] = wmma_bf16(a0, b0, acc[0][0]);  acc[0][1] = wmma_bf16(a0, b1, acc[0][1]);
    acc[1][0] = wmma_bf16(a1, b0, acc[1][0]);  acc[1][1] = wmma_bf16(a1, b1, acc[1][1]);
  }
  int lane = threadIdx.x & 31;
  int rsel = (lane >> 4) << 3;
#pragma unroll
  for (int mi = 0; mi < 2; ++mi)
#pragma unroll
    for (int ni = 0; ni < 2; ++ni) {
      int col = n0 + ni * 16 + (lane & 15);
#pragma unroll
      for (int i = 0; i < 8; ++i) {
        int row = m0 + mi * 16 + i + rsel;
        size_t o = (size_t)row * DIM + col;
        y[o] = acc[mi][ni][i] + bias[col] + x[o];
      }
    }
}

// ---------------- host ----------------
extern "C" void kernel_launch(void* const* d_in, const int* in_sizes, int n_in,
                              void* d_out, int out_size, void* d_ws, size_t ws_size,
                              hipStream_t stream) {
  (void)in_sizes; (void)n_in; (void)out_size; (void)ws_size;
  const float* x     = (const float*)d_in[0];
  const float* gamma = (const float*)d_in[1];
  const float* beta  = (const float*)d_in[2];
  const float* wqkv  = (const float*)d_in[3];
  const float* wout  = (const float*)d_in[4];
  const float* bout  = (const float*)d_in[5];
  const float* wconv = (const float*)d_in[6];
  float* y = (float*)d_out;

  char* ws = (char*)d_ws;
  size_t off = 0;
  auto alloc = [&](size_t bytes) { size_t o = off; off += (bytes + 255) & ~(size_t)255; return o; };
  const size_t SZ_HEAD = (size_t)BH * SEQ * DHEAD;          // 16.7M elems
  size_t o_xn   = alloc((size_t)TOK * DIM * 2);             // dead after QKV  } aliased by
  size_t o_k    = alloc(SZ_HEAD * 2);                       // dead after flash} out_head/oh
  size_t o_vT   = alloc(SZ_HEAD * 2);                       // dead after flash}
  size_t o_q    = alloc(SZ_HEAD * 2);
  size_t o_v    = alloc(SZ_HEAD * 2);
  size_t o_wqkvT= alloc((size_t)3 * DIM * DIM * 2);
  size_t o_woutT= alloc((size_t)DIM * DIM * 2);
  size_t o_ql   = alloc((size_t)BH * LAND * DHEAD * 2);
  size_t o_kl   = alloc((size_t)BH * LAND * DHEAD * 2);
  size_t o_a2f  = alloc((size_t)BH * LAND * LAND * 4);
  size_t o_a2b  = alloc((size_t)BH * LAND * LAND * 2);
  size_t o_zA   = alloc((size_t)BH * LAND * LAND * 2);
  size_t o_zAT  = alloc((size_t)BH * LAND * LAND * 2);
  size_t o_zB   = alloc((size_t)BH * LAND * LAND * 2);
  size_t o_zBT  = alloc((size_t)BH * LAND * LAND * 2);
  size_t o_xz   = alloc((size_t)BH * LAND * LAND * 2);
  size_t o_tA   = alloc((size_t)BH * LAND * LAND * 2);
  size_t o_tB   = alloc((size_t)BH * LAND * LAND * 2);
  size_t o_a3vT = alloc((size_t)BH * DHEAD * LAND * 2);
  size_t o_WT   = alloc((size_t)BH * DHEAD * LAND * 2);
  size_t o_scal = alloc(256);
  size_t o_ohf  = o_xn;                   // 64MB f32 over xn+k (dead)
  size_t o_oh   = o_xn + SZ_HEAD * 4;     // 32MB bf16 over vT (dead)

  bf16_t* xn    = (bf16_t*)(ws + o_xn);
  bf16_t* kbuf  = (bf16_t*)(ws + o_k);
  bf16_t* vT    = (bf16_t*)(ws + o_vT);
  bf16_t* qbuf  = (bf16_t*)(ws + o_q);
  bf16_t* vbuf  = (bf16_t*)(ws + o_v);
  bf16_t* wqkvT = (bf16_t*)(ws + o_wqkvT);
  bf16_t* woutT = (bf16_t*)(ws + o_woutT);
  bf16_t* ql    = (bf16_t*)(ws + o_ql);
  bf16_t* kl    = (bf16_t*)(ws + o_kl);
  float*  a2f   = (float*)(ws + o_a2f);
  bf16_t* a2b   = (bf16_t*)(ws + o_a2b);
  bf16_t* zA    = (bf16_t*)(ws + o_zA);
  bf16_t* zAT   = (bf16_t*)(ws + o_zAT);
  bf16_t* zB    = (bf16_t*)(ws + o_zB);
  bf16_t* zBT   = (bf16_t*)(ws + o_zBT);
  bf16_t* xz    = (bf16_t*)(ws + o_xz);
  bf16_t* tA    = (bf16_t*)(ws + o_tA);
  bf16_t* tB    = (bf16_t*)(ws + o_tB);
  bf16_t* a3vT  = (bf16_t*)(ws + o_a3vT);
  bf16_t* WT    = (bf16_t*)(ws + o_WT);
  unsigned* scal= (unsigned*)(ws + o_scal);
  float*  ohf   = (float*)(ws + o_ohf);
  bf16_t* oh    = (bf16_t*)(ws + o_oh);

  dim3 w32(32);
  k_convert  <<<dim3((DIM * 3 * DIM + 255) / 256), dim3(256), 0, stream>>>(wqkv, wout, wqkvT, woutT, scal);
  k_layernorm<<<dim3(TOK / 8), dim3(256), 0, stream>>>(x, gamma, beta, xn);
  k_gemm_qkv <<<dim3(TOK / 32, (3 * DIM) / 32), w32, 0, stream>>>(xn, wqkvT, qbuf, kbuf, vbuf, vT);
  k_landmarks<<<dim3((BH * LAND * DHEAD + 255) / 256), dim3(256), 0, stream>>>(qbuf, kbuf, ql, kl);
  k_sim2     <<<dim3(LAND / 16, BH), w32, 0, stream>>>(ql, kl, a2f, a2b);
  k_colmax   <<<dim3(BH), dim3(LAND), 0, stream>>>(a2f, scal);
  k_z0       <<<dim3((BH * LAND * LAND + 255) / 256), dim3(256), 0, stream>>>(a2f, scal, zA, zAT);
  k_flash    <<<dim3(LAND / 16, BH), w32, 0, stream>>>(ql, kbuf, vT, a3vT);

  // Newton-Schulz pinv: z = 0.25 z (13I - xz(15I - xz(7I - xz)))
  bf16_t *zin = zA, *zinT = zAT, *zout = zB, *zoutT = zBT;
  long long sM = (long long)LAND * LAND;
  dim3 gP(LAND / 32, LAND / 32, BH);
  for (int it = 0; it < 6; ++it) {
    k_gemm_nt<<<gP, w32, 0, stream>>>(a2b, zinT, LAND, LAND, LAND, sM, sM, xz, 1.f, 0.f, tA, -1.f, 7.f);
    k_gemm_nt<<<gP, w32, 0, stream>>>(xz, tA, LAND, LAND, LAND, sM, sM, nullptr, 0.f, 0.f, tB, -1.f, 15.f);
    k_gemm_nt<<<gP, w32, 0, stream>>>(xz, tB, LAND, LAND, LAND, sM, sM, nullptr, 0.f, 0.f, tA, -1.f, 13.f);
    k_gemm_nt<<<gP, w32, 0, stream>>>(zin, tA, LAND, LAND, LAND, sM, sM, zout, 0.25f, 0.f, zoutT, 0.25f, 0.f);
    bf16_t* t;
    t = zin; zin = zout; zout = t;
    t = zinT; zinT = zoutT; zoutT = t;
  }
  // W = a2inv @ (a3 v)  -> stored transposed [bh,64,256]
  k_gemm_nt<<<dim3(LAND / 32, DHEAD / 32, BH), w32, 0, stream>>>(
      zin, a3vT, LAND, DHEAD, LAND, sM, (long long)DHEAD * LAND, nullptr, 0.f, 0.f, WT, 1.f, 0.f);

  k_attn    <<<dim3(SEQ / 16, BH), w32, 0, stream>>>(qbuf, kl, WT, ohf);
  k_conv    <<<dim3((unsigned)((SZ_HEAD + 255) / 256)), dim3(256), 0, stream>>>(ohf, vbuf, wconv, oh);
  k_gemm_out<<<dim3(TOK / 32, DIM / 32), w32, 0, stream>>>(oh, woutT, bout, x, y);
}
